// MixLoraSparseMoe_33913061769875
// MI455X (gfx1250) — compile-verified, matrix operand order checked by reading, not run on hardware
//
#include <hip/hip_runtime.h>
#include <cstdint>
#include <cstddef>

typedef __bf16 bf16;
typedef __bf16 v16bf __attribute__((ext_vector_type(16)));
typedef __bf16 v8bf  __attribute__((ext_vector_type(8)));
typedef __bf16 v4bf  __attribute__((ext_vector_type(4)));
typedef float  v8f   __attribute__((ext_vector_type(8)));

#define DEV_INLINE __device__ __forceinline__

namespace {
constexpr int kT  = 2048;   // tokens
constexpr int kH  = 2048;   // hidden
constexpr int kF  = 8192;   // ffn
constexpr int kE  = 8;      // experts
constexpr int kR  = 16;     // lora rank
constexpr int kTT = 2 * kT; // routed rows (top_k = 2)
constexpr float kScale = 2.0f; // LORA_SCALING
}

// ---------------------------------------------------------------------------
// gfx1250 async global -> LDS (16B per lane), tracked by ASYNCcnt.
// GVS mode: global addr = SGPR base + 32-bit VGPR offset; LDS addr from VDST VGPR.
// ---------------------------------------------------------------------------
DEV_INLINE void async_copy_b128(unsigned lds_byte_off, unsigned gbyte_off,
                                const void* sbase) {
  asm volatile("global_load_async_to_lds_b128 %0, %1, %2"
               :: "v"(lds_byte_off), "v"(gbyte_off), "s"(sbase)
               : "memory");
}
DEV_INLINE void wait_async0() {
  asm volatile("s_wait_asynccnt 0x0" ::: "memory");
}
// Low 32 bits of a generic LDS pointer are the LDS byte offset
template <typename T>
DEV_INLINE unsigned lds_off_u32(const T* p) {
  return (unsigned)(unsigned long long)(unsigned long long)(uintptr_t)p;
}

// ---------------------------------------------------------------------------
// WMMA fragment load from a row-major LDS tile (rows x BK bf16, BK = ld).
// ISA 16-bit A-matrix 16x32 layout; also valid as the B fragment for
// x @ W.T when W's row-major tile is staged.
// ---------------------------------------------------------------------------
DEV_INLINE v16bf load_frag(const bf16* base, int ld, int lane) {
  const int r = lane & 15, half = lane >> 4;
  const bf16* p = base + r * ld + half * 8;
  v8bf lo = *(const v8bf*)(p);        // K = half*8 .. +7
  v8bf hi = *(const v8bf*)(p + 16);   // K = 16 + half*8 .. +7
  v16bf v;
  #pragma unroll
  for (int i = 0; i < 8; ++i) { v[i] = lo[i]; v[i + 8] = hi[i]; }
  return v;
}

// ---------------------------------------------------------------------------
// 0) f32 -> bf16 cast (4 elems / thread)
// ---------------------------------------------------------------------------
__global__ void cast_bf16_kernel(const float* __restrict__ src,
                                 bf16* __restrict__ dst, long n) {
  const long i = ((long)blockIdx.x * blockDim.x + threadIdx.x) * 4;
  if (i + 3 < n) {
    float4 v = *(const float4*)(src + i);
    v4bf o;
    o[0] = (bf16)v.x; o[1] = (bf16)v.y; o[2] = (bf16)v.z; o[3] = (bf16)v.w;
    *(v4bf*)(dst + i) = o;
  }
}

// ---------------------------------------------------------------------------
// 1) Router: wave-per-token logits -> softmax -> top-2 -> renormalize
// ---------------------------------------------------------------------------
__global__ __launch_bounds__(256) void router_kernel(
    const float* __restrict__ h, const float* __restrict__ gw,
    int* __restrict__ sel, float* __restrict__ wts) {
  const int lane = threadIdx.x & 31, wave = threadIdx.x >> 5;
  const int t = blockIdx.x * 8 + wave;
  float lg[kE];
  #pragma unroll
  for (int e = 0; e < kE; ++e) {
    float s = 0.f;
    for (int k = lane; k < kH; k += 32)
      s += h[(size_t)t * kH + k] * gw[(size_t)e * kH + k];
    #pragma unroll
    for (int o = 16; o > 0; o >>= 1) s += __shfl_xor(s, o, 32);
    lg[e] = s; // identical across lanes after xor-reduce
  }
  if (lane == 0) {
    float mx = lg[0];
    #pragma unroll
    for (int e = 1; e < kE; ++e) mx = fmaxf(mx, lg[e]);
    float p[kE];
    #pragma unroll
    for (int e = 0; e < kE; ++e) p[e] = __expf(lg[e] - mx);
    int i0 = 0;
    #pragma unroll
    for (int e = 1; e < kE; ++e) if (p[e] > p[i0]) i0 = e;
    int i1 = (i0 == 0) ? 1 : 0;
    #pragma unroll
    for (int e = 0; e < kE; ++e) if (e != i0 && p[e] > p[i1]) i1 = e;
    const float wsum = p[i0] + p[i1];
    sel[t * 2 + 0] = i0; wts[t * 2 + 0] = p[i0] / wsum;
    sel[t * 2 + 1] = i1; wts[t * 2 + 1] = p[i1] / wsum;
  }
}

// ---------------------------------------------------------------------------
// 2) Build per-expert gathered row lists (serial, deterministic)
// ---------------------------------------------------------------------------
__global__ void build_lists_kernel(const int* __restrict__ sel,
                                   int* __restrict__ row2tok,
                                   int* __restrict__ tok2row,
                                   int* __restrict__ off) {
  if (threadIdx.x != 0 || blockIdx.x != 0) return;
  int cnt[kE];
  for (int e = 0; e < kE; ++e) cnt[e] = 0;
  for (int i = 0; i < kTT; ++i) cnt[sel[i]]++;
  int o[kE + 1];
  o[0] = 0;
  for (int e = 0; e < kE; ++e) o[e + 1] = o[e] + cnt[e];
  for (int e = 0; e <= kE; ++e) off[e] = o[e];
  int cur[kE];
  for (int e = 0; e < kE; ++e) cur[e] = o[e];
  for (int t = 0; t < kT; ++t)
    for (int k = 0; k < 2; ++k) {
      const int e = sel[t * 2 + k];
      const int r = cur[e]++;
      row2tok[r] = t;
      tok2row[t * 2 + k] = r;
    }
}

// ---------------------------------------------------------------------------
// 3) LoRA A-projections: a_g[row,:] = h[tok] @ Ag[e].T, a_u likewise.
// ---------------------------------------------------------------------------
__global__ __launch_bounds__(256) void aproj_kernel(
    const float* __restrict__ h, const float* __restrict__ Ag,
    const float* __restrict__ Au, const int* __restrict__ row2tok,
    const int* __restrict__ off, float* __restrict__ ag,
    float* __restrict__ au) {
  constexpr int RM = 32, CK = 128;
  __shared__ float shh[RM * CK];
  __shared__ float shA[32 * CK];
  __shared__ int toks[RM];
  const int e = blockIdx.y;
  const int rbeg = off[e], rend = off[e + 1];
  const int m0 = rbeg + blockIdx.x * RM;
  if (m0 >= rend) return;
  const int tid = threadIdx.x;
  if (tid < RM) {
    int rr = m0 + tid; if (rr > rend - 1) rr = rend - 1;
    toks[tid] = row2tok[rr];
  }
  __syncthreads();
  const int rl = tid & 31, grp = tid >> 5;
  float acc[4] = {0.f, 0.f, 0.f, 0.f};
  for (int k0 = 0; k0 < kH; k0 += CK) {
    for (int i = tid; i < RM * CK; i += 256) {
      const int r = i >> 7, k = i & (CK - 1);
      shh[i] = h[(size_t)toks[r] * kH + k0 + k];
    }
    for (int i = tid; i < 32 * CK; i += 256) {
      const int o = i >> 7, k = i & (CK - 1);
      shA[i] = (o < kR) ? Ag[((size_t)e * kR + o) * kH + k0 + k]
                        : Au[((size_t)e * kR + (o - kR)) * kH + k0 + k];
    }
    __syncthreads();
    for (int k = 0; k < CK; ++k) {
      const float hv = shh[rl * CK + k];
      #pragma unroll
      for (int j = 0; j < 4; ++j) acc[j] += hv * shA[(grp + 8 * j) * CK + k];
    }
    __syncthreads();
  }
  const int row = m0 + rl;
  if (row < rend) {
    #pragma unroll
    for (int j = 0; j < 4; ++j) {
      const int o = grp + 8 * j;
      if (o < kR) ag[(size_t)row * kR + o] = acc[j];
      else        au[(size_t)row * kR + (o - kR)] = acc[j];
    }
  }
}

// ---------------------------------------------------------------------------
// 4) Fused gate/up GEMM (bf16 WMMA) + rank-16 LoRA + SiLU -> act bf16 [2T,F]
//    128x128 tile, K-step 32, double-buffered LDS fed by async global->LDS.
// ---------------------------------------------------------------------------
__global__ __launch_bounds__(256) void gateup_wmma_kernel(
    const bf16* __restrict__ hB, const bf16* __restrict__ WgB,
    const bf16* __restrict__ WuB, const float* __restrict__ Bg,
    const float* __restrict__ Bu, const float* __restrict__ ag,
    const float* __restrict__ au, const int* __restrict__ row2tok,
    const int* __restrict__ off, bf16* __restrict__ act) {
  constexpr int TM = 128, TN = 128, BK = 32;
  __shared__ __align__(16) bf16 sA[2][TM * BK];
  __shared__ __align__(16) bf16 sG[2][TN * BK];
  __shared__ __align__(16) bf16 sU[2][TN * BK];
  __shared__ int toks[TM];
  __shared__ float sag[TM * kR], sau[TM * kR], sbg[TN * kR], sbu[TN * kR];

  const int e = blockIdx.z;
  const int rbeg = off[e], rend = off[e + 1];
  const int m0 = rbeg + blockIdx.y * TM;
  if (m0 >= rend) return;
  const int f0 = blockIdx.x * TN;
  const int tid = threadIdx.x, lane = tid & 31, wave = tid >> 5;

  if (tid < TM) {
    int rr = m0 + tid; if (rr > rend - 1) rr = rend - 1;
    toks[tid] = row2tok[rr];
  }
  __syncthreads();

  const unsigned oA[2] = {lds_off_u32(sA[0]), lds_off_u32(sA[1])};
  const unsigned oG[2] = {lds_off_u32(sG[0]), lds_off_u32(sG[1])};
  const unsigned oU[2] = {lds_off_u32(sU[0]), lds_off_u32(sU[1])};

  auto stage = [&](int buf, int k0) {
    for (int ch = tid; ch < (TM * BK) / 8; ch += 256) { // 512 chunks of 16B
      const int r = ch >> 2, c8 = (ch & 3) << 3;
      const unsigned l = (unsigned)((r * BK + c8) * 2);
      async_copy_b128(oA[buf] + l,
                      (unsigned)(((size_t)toks[r] * kH + k0 + c8) * 2), hB);
      async_copy_b128(oG[buf] + l,
                      (unsigned)(((size_t)(f0 + r) * kH + k0 + c8) * 2), WgB);
      async_copy_b128(oU[buf] + l,
                      (unsigned)(((size_t)(f0 + r) * kH + k0 + c8) * 2), WuB);
    }
  };

  v8f accg[8], accu[8];
  #pragma unroll
  for (int n = 0; n < 8; ++n) {
    accg[n] = v8f{0.f, 0.f, 0.f, 0.f, 0.f, 0.f, 0.f, 0.f};
    accu[n] = accg[n];
  }

  stage(0, 0);
  wait_async0();
  __syncthreads();

  for (int k0 = 0, it = 0; k0 < kH; k0 += BK, ++it) {
    const int cur = it & 1;
    if (k0 + BK < kH) stage(cur ^ 1, k0 + BK); // overlap with WMMAs below
    const bf16* pA = sA[cur];
    const bf16* pG = sG[cur];
    const bf16* pU = sU[cur];
    const v16bf afrag = load_frag(pA + (wave * 16) * BK, BK, lane);
    v16bf bg = load_frag(pG, BK, lane);
    v16bf bu = load_frag(pU, BK, lane);
    #pragma unroll
    for (int n = 0; n < 8; ++n) {
      v16bf bgn = bg, bun = bu;
      if (n < 7) { // prefetch next fragments before issuing WMMAs
        bgn = load_frag(pG + (n + 1) * 16 * BK, BK, lane);
        bun = load_frag(pU + (n + 1) * 16 * BK, BK, lane);
      }
      accg[n] = __builtin_amdgcn_wmma_f32_16x16x32_bf16(
          false, afrag, false, bg, (short)0, accg[n], false, false);
      accu[n] = __builtin_amdgcn_wmma_f32_16x16x32_bf16(
          false, afrag, false, bu, (short)0, accu[n], false, false);
      bg = bgn; bu = bun;
    }
    wait_async0();
    __syncthreads();
  }

  // Epilogue: + LORA_SCALING * (a @ B.T), SiLU(g) * u, store bf16.
  for (int i = tid; i < TM * kR; i += 256) {
    const int r = i >> 4, j = i & 15;
    int rr = m0 + r; if (rr > rend - 1) rr = rend - 1;
    sag[i] = ag[(size_t)rr * kR + j];
    sau[i] = au[(size_t)rr * kR + j];
    sbg[i] = Bg[((size_t)e * kF + f0 + r) * kR + j];
    sbu[i] = Bu[((size_t)e * kF + f0 + r) * kR + j];
  }
  __syncthreads();
  const int mbase = wave * 16 + ((lane >> 4) << 3);
  const int nl = lane & 15;
  #pragma unroll
  for (int n = 0; n < 8; ++n) {
    const int nc = n * 16 + nl;
    #pragma unroll
    for (int i = 0; i < 8; ++i) {
      const int ml = mbase + i;
      const int row = m0 + ml;
      float g = accg[n][i], u = accu[n][i];
      float dg = 0.f, du = 0.f;
      #pragma unroll
      for (int r = 0; r < kR; ++r) {
        dg += sag[ml * kR + r] * sbg[nc * kR + r];
        du += sau[ml * kR + r] * sbu[nc * kR + r];
      }
      g += kScale * dg;
      u += kScale * du;
      const float silu = g / (1.f + __expf(-g));
      if (row < rend) act[(size_t)row * kF + f0 + nc] = (bf16)(silu * u);
    }
  }
}

// ---------------------------------------------------------------------------
// 5) Down GEMM (bf16 WMMA) with fused act@Ad.T (one extra WMMA / k-step)
//    + rank-16 LoRA epilogue -> y f32 [2T, H]; async double-buffered LDS.
// ---------------------------------------------------------------------------
__global__ __launch_bounds__(256) void down_wmma_kernel(
    const bf16* __restrict__ actB, const bf16* __restrict__ WdB,
    const bf16* __restrict__ AdB, const float* __restrict__ Bd,
    const int* __restrict__ off, float* __restrict__ y) {
  constexpr int TM = 128, TN = 128, BK = 32;
  __shared__ __align__(16) bf16 sA[2][TM * BK];
  __shared__ __align__(16) bf16 sW[2][TN * BK];
  __shared__ __align__(16) bf16 sAd[2][kR * BK];
  __shared__ float sad[TM * kR];
  __shared__ float sbd[TN * kR];

  const int e = blockIdx.z;
  const int rbeg = off[e], rend = off[e + 1];
  const int m0 = rbeg + blockIdx.y * TM;
  if (m0 >= rend) return;
  const int n0 = blockIdx.x * TN;
  const int tid = threadIdx.x, lane = tid & 31, wave = tid >> 5;

  const unsigned oA[2] = {lds_off_u32(sA[0]), lds_off_u32(sA[1])};
  const unsigned oW[2] = {lds_off_u32(sW[0]), lds_off_u32(sW[1])};
  const unsigned oD[2] = {lds_off_u32(sAd[0]), lds_off_u32(sAd[1])};

  auto stage = [&](int buf, int k0) {
    for (int ch = tid; ch < (TM * BK) / 8; ch += 256) {
      const int r = ch >> 2, c8 = (ch & 3) << 3;
      int rr = m0 + r; if (rr > rend - 1) rr = rend - 1;
      const unsigned l = (unsigned)((r * BK + c8) * 2);
      async_copy_b128(oA[buf] + l,
                      (unsigned)(((size_t)rr * kF + k0 + c8) * 2), actB);
      async_copy_b128(oW[buf] + l,
                      (unsigned)(((size_t)(n0 + r) * kF + k0 + c8) * 2), WdB);
    }
    if (tid < (kR * BK) / 8) { // 64 chunks for the Ad tile
      const int r = tid >> 2, c8 = (tid & 3) << 3;
      async_copy_b128(oD[buf] + (unsigned)((r * BK + c8) * 2),
                      (unsigned)((((size_t)e * kR + r) * kF + k0 + c8) * 2),
                      AdB);
    }
  };

  v8f acc[8], accad;
  #pragma unroll
  for (int n = 0; n < 8; ++n)
    acc[n] = v8f{0.f, 0.f, 0.f, 0.f, 0.f, 0.f, 0.f, 0.f};
  accad = acc[0];

  stage(0, 0);
  wait_async0();
  __syncthreads();

  for (int k0 = 0, it = 0; k0 < kF; k0 += BK, ++it) {
    const int cur = it & 1;
    if (k0 + BK < kF) stage(cur ^ 1, k0 + BK);
    const bf16* pA = sA[cur];
    const bf16* pW = sW[cur];
    const v16bf afrag = load_frag(pA + (wave * 16) * BK, BK, lane);
    v16bf b = load_frag(pW, BK, lane);
    #pragma unroll
    for (int n = 0; n < 8; ++n) {
      v16bf bn = b;
      if (n < 7) bn = load_frag(pW + (n + 1) * 16 * BK, BK, lane);
      acc[n] = __builtin_amdgcn_wmma_f32_16x16x32_bf16(
          false, afrag, false, b, (short)0, acc[n], false, false);
      b = bn;
    }
    v16bf adf = load_frag(sAd[cur], BK, lane);
    accad = __builtin_amdgcn_wmma_f32_16x16x32_bf16(
        false, afrag, false, adf, (short)0, accad, false, false);
    wait_async0();
    __syncthreads();
  }

  // Spill per-row ad (act @ Ad.T) to LDS so every lane can read its row.
  {
    const int mb = wave * 16 + ((lane >> 4) << 3);
    const int rl = lane & 15;
    #pragma unroll
    for (int i = 0; i < 8; ++i) sad[(mb + i) * kR + rl] = accad[i];
  }
  for (int i = tid; i < TN * kR; i += 256) {
    const int r = i >> 4, j = i & 15;
    sbd[i] = Bd[((size_t)e * kH + n0 + r) * kR + j];
  }
  __syncthreads();

  const int mbase = wave * 16 + ((lane >> 4) << 3);
  const int nl = lane & 15;
  #pragma unroll
  for (int n = 0; n < 8; ++n) {
    const int nc = n * 16 + nl;
    #pragma unroll
    for (int i = 0; i < 8; ++i) {
      const int ml = mbase + i, row = m0 + ml;
      float v = acc[n][i];
      float d = 0.f;
      #pragma unroll
      for (int r = 0; r < kR; ++r) d += sad[ml * kR + r] * sbd[nc * kR + r];
      v += kScale * d;
      if (row < rend) y[(size_t)row * kH + n0 + nc] = v;
    }
  }
}

// ---------------------------------------------------------------------------
// 6) Combine: out[t,:] = w0 * y[row0,:] + w1 * y[row1,:]
// ---------------------------------------------------------------------------
__global__ void combine_kernel(const float* __restrict__ y,
                               const float* __restrict__ wts,
                               const int* __restrict__ tok2row,
                               float* __restrict__ out) {
  const long idx = (long)blockIdx.x * 256 + threadIdx.x; // T*H total
  const int t = (int)(idx >> 11);
  const int c = (int)(idx & (kH - 1));
  const int r0 = tok2row[t * 2 + 0], r1 = tok2row[t * 2 + 1];
  out[idx] = wts[t * 2 + 0] * y[(size_t)r0 * kH + c] +
             wts[t * 2 + 1] * y[(size_t)r1 * kH + c];
}

// ---------------------------------------------------------------------------
// Launch
// ---------------------------------------------------------------------------
extern "C" void kernel_launch(void* const* d_in, const int* in_sizes, int n_in,
                              void* d_out, int out_size, void* d_ws,
                              size_t ws_size, hipStream_t stream) {
  (void)in_sizes; (void)n_in; (void)out_size; (void)ws_size;
  const float* hs = (const float*)d_in[0];
  const float* gw = (const float*)d_in[1];
  const float* Wg = (const float*)d_in[2];
  const float* Wu = (const float*)d_in[3];
  const float* Wd = (const float*)d_in[4];
  const float* Ag = (const float*)d_in[5];
  const float* Bg = (const float*)d_in[6];
  const float* Au = (const float*)d_in[7];
  const float* Bu = (const float*)d_in[8];
  const float* Ad = (const float*)d_in[9];
  const float* Bd = (const float*)d_in[10];
  float* out = (float*)d_out;

  char* ws = (char*)d_ws;
  size_t cur = 0;
  auto take = [&](size_t bytes) -> char* {
    char* p = ws + cur;
    cur += (bytes + 255) & ~(size_t)255;
    return p;
  };
  bf16* hB  = (bf16*)take((size_t)kT * kH * 2);
  bf16* WgB = (bf16*)take((size_t)kF * kH * 2);
  bf16* WuB = (bf16*)take((size_t)kF * kH * 2);
  bf16* WdB = (bf16*)take((size_t)kH * kF * 2);
  bf16* AdB = (bf16*)take((size_t)kE * kR * kF * 2);
  int*   sel     = (int*)take((size_t)kTT * 4);
  float* wts     = (float*)take((size_t)kTT * 4);
  int*   row2tok = (int*)take((size_t)kTT * 4);
  int*   tok2row = (int*)take((size_t)kTT * 4);
  int*   off     = (int*)take((size_t)(kE + 1) * 4);
  float* ag  = (float*)take((size_t)kTT * kR * 4);
  float* au  = (float*)take((size_t)kTT * kR * 4);
  bf16*  act = (bf16*)take((size_t)kTT * kF * 2);
  float* yb  = (float*)take((size_t)kTT * kH * 4);

  auto cast = [&](const float* s, bf16* d, long n) {
    cast_bf16_kernel<<<dim3((unsigned)((n / 4 + 255) / 256)), dim3(256), 0,
                       stream>>>(s, d, n);
  };
  cast(hs, hB, (long)kT * kH);
  cast(Wg, WgB, (long)kF * kH);
  cast(Wu, WuB, (long)kF * kH);
  cast(Wd, WdB, (long)kH * kF);
  cast(Ad, AdB, (long)kE * kR * kF);

  router_kernel<<<dim3(kT / 8), dim3(256), 0, stream>>>(hs, gw, sel, wts);
  build_lists_kernel<<<dim3(1), dim3(1), 0, stream>>>(sel, row2tok, tok2row,
                                                      off);
  aproj_kernel<<<dim3(kTT / 32, kE), dim3(256), 0, stream>>>(
      hs, Ag, Au, row2tok, off, ag, au);
  gateup_wmma_kernel<<<dim3(kF / 128, kTT / 128, kE), dim3(256), 0, stream>>>(
      hB, WgB, WuB, Bg, Bu, ag, au, row2tok, off, act);
  down_wmma_kernel<<<dim3(kH / 128, kTT / 128, kE), dim3(256), 0, stream>>>(
      act, WdB, AdB, Bd, off, yb);
  combine_kernel<<<dim3((kT * kH) / 256), dim3(256), 0, stream>>>(
      yb, wts, tok2row, out);
}
